// CausalSelfAttention_77283641524841
// MI455X (gfx1250) — matrix operand order repaired and checked by reading, not compiled, not run on hardware
//
#include <hip/hip_runtime.h>
#include <hip/hip_bf16.h>
#include <stdint.h>

// CausalSelfAttention on MI455X (gfx1250, wave32).
//   1) convert X f32->bf16; transpose+convert W_qkv, W_proj (bf16, N-major)
//   2) QKV GEMM: TDM (tensor_load_to_lds) double-buffered staging, wmma bf16;
//      epilogue scatters Q,K -> [bh][T][64], V -> [bh][64][T]
//   3) flash attention (online softmax, causal), wmma bf16
//   4) out-projection GEMM -> f32
// WMMA fragments are 16B-contiguous loads:
//   A-frag lane(g,m): A[m][8g..8g+7] ++ A[m][16+8g..+7]   (row-major A)
//   B-frag lane(g,n): Bt[n][8g..8g+7] ++ Bt[n][16+8g..+7] (Bt = N x K)

typedef unsigned short u16;
typedef unsigned int   u32;
typedef __attribute__((ext_vector_type(4)))  u32    v4u;
typedef __attribute__((ext_vector_type(8)))  int    v8i;
typedef __attribute__((ext_vector_type(4)))  int    v4i;
typedef __attribute__((ext_vector_type(16))) __bf16 v16bf;
typedef __attribute__((ext_vector_type(8)))  float  v8f;

union Frag {
    v16bf v;
    u16   s[16];
    v4u   q[2];
};

__device__ __forceinline__ u16 f2bf(float f) {
    union { float f; u32 u; } x; x.f = f;
    u32 u = x.u;
    u += 0x7FFFu + ((u >> 16) & 1u);
    return (u16)(u >> 16);
}

__device__ __forceinline__ v8f wmma_bf16(const Frag& a, const Frag& b, v8f c) {
    return __builtin_amdgcn_wmma_f32_16x16x32_bf16(
        false, a.v, false, b.v, (short)0, c, false, false);
}

// ---------------------------------------------------------------------------
// TDM: DMA a 2D tile (tile_rows x 32 bf16, row stride Kd) from global to LDS.
// D# per CDNA5 ISA ch.8: group0 {count, lds_addr, global_addr, type=2},
// group1 {data_size=2B, tensor_dim0/1, tile_dim0=32, tile_dim1, dim0_stride}.
// LDS byte offset = low 32 bits of the generic pointer (LDS aperture rule).
// Toolchain uses the 6-arg builtin: (v4u, v8i, v4i, v4i, v8i, i32 cpol).
// ---------------------------------------------------------------------------
__device__ __forceinline__ void tdm_load_tile(const u16* gsrc, u16* lds_dst,
                                              int tensor_rows, int Kd,
                                              int tile_rows) {
    unsigned long long ga = (unsigned long long)(uintptr_t)gsrc;
    v4u g0;
    g0[0] = 1u;                                    // count=1, user descriptor
    g0[1] = (u32)(uintptr_t)lds_dst;               // lds_addr (bytes)
    g0[2] = (u32)ga;                               // global_addr[31:0]
    g0[3] = (u32)(ga >> 32) | 0x80000000u;         // addr[56:32] | type=2
    v8i g1;
    g1[0] = (int)(1u << 16);                       // data_size = 1 -> 2 bytes
    g1[1] = (int)((u32)Kd << 16);                  // tensor_dim0 low 16
    g1[2] = (int)(((u32)Kd >> 16) | ((u32)tensor_rows << 16));  // dim0 hi | dim1 lo
    g1[3] = (int)(((u32)tensor_rows >> 16) | (32u << 16));      // dim1 hi | tile_dim0=32
    g1[4] = tile_rows;                             // tile_dim1 (tile_dim2=0)
    g1[5] = Kd;                                    // tensor_dim0_stride low 32
    g1[6] = 0;
    g1[7] = 0;
    v4i gz4 = {0, 0, 0, 0};
    v8i gz8 = {0, 0, 0, 0, 0, 0, 0, 0};
    __builtin_amdgcn_tensor_load_to_lds(g0, g1, gz4, gz4, gz8, 0);
}

// ---------------------------------------------------------------------------
// f32 -> bf16 convert
// ---------------------------------------------------------------------------
__global__ void f32_to_bf16_kernel(const float* __restrict__ in,
                                   u16* __restrict__ out, int n) {
    int i = blockIdx.x * blockDim.x + threadIdx.x;
    if (i < n) out[i] = f2bf(in[i]);
}

// ---------------------------------------------------------------------------
// Tiled transpose + convert: in [R][Ccols] f32 -> out [Ccols][R] bf16
// ---------------------------------------------------------------------------
__global__ __launch_bounds__(256)
void transpose_convert_kernel(const float* __restrict__ in,
                              u16* __restrict__ out, int R, int Ccols) {
    __shared__ float t[32][33];
    const int c0 = blockIdx.x * 32;
    const int r0 = blockIdx.y * 32;
    const int x = threadIdx.x;
    const int y = threadIdx.y;
#pragma unroll
    for (int i = 0; i < 32; i += 8)
        t[y + i][x] = in[(size_t)(r0 + y + i) * Ccols + c0 + x];
    __syncthreads();
#pragma unroll
    for (int i = 0; i < 32; i += 8)
        out[(size_t)(c0 + y + i) * R + r0 + x] = f2bf(t[x][y + i]);
}

// ---------------------------------------------------------------------------
// bf16 GEMM, TN form: C[M,N] = A[M,K] * Bt[N,K]^T.
// Block = 128 threads = 4 waves; block tile 128x64; wave tile 32x64; K-step 32.
// Staging: TDM tensor_load_to_lds, double-buffered (wave 0 issues + waits
// TENSORcnt, workgroup barriers publish). 8 wmma / wave / K-step.
// mode==0: f32 row-major. mode==1: QKV scatter (V stored d-major, packed b128).
// ---------------------------------------------------------------------------
__global__ __launch_bounds__(128)
void gemm_bf16_tn_kernel(const u16* __restrict__ Am, const u16* __restrict__ Bt,
                         int M, int N, int Kd,
                         float* __restrict__ outF,
                         u16* __restrict__ oQ, u16* __restrict__ oK,
                         u16* __restrict__ oVt, int mode) {
    __shared__ u16 la[2][128][32];
    __shared__ u16 lbT[2][64][32];

    const int tid  = threadIdx.x;
    const int wave = tid >> 5;
    const int lane = tid & 31;
    const int g    = lane >> 4;
    const int ln   = lane & 15;
    const int tileM = blockIdx.y * 128;
    const int tileN = blockIdx.x * 64;

    const u16* Abase = Am + (size_t)tileM * Kd;
    const u16* Bbase = Bt + (size_t)tileN * Kd;
    const int nk = Kd >> 5;

    v8f acc[2][4] = {};

    if (wave == 0) {
        tdm_load_tile(Abase, &la[0][0][0], M, Kd, 128);
        tdm_load_tile(Bbase, &lbT[0][0][0], N, Kd, 64);
    }

    for (int ki = 0; ki < nk; ++ki) {
        const int cur = ki & 1;
        if (ki + 1 < nk) {
            if (wave == 0) {
                tdm_load_tile(Abase + (ki + 1) * 32, &la[cur ^ 1][0][0], M, Kd, 128);
                tdm_load_tile(Bbase + (ki + 1) * 32, &lbT[cur ^ 1][0][0], N, Kd, 64);
                __builtin_amdgcn_s_wait_tensorcnt(2);   // tile ki's 2 loads done
            }
        } else if (wave == 0) {
            __builtin_amdgcn_s_wait_tensorcnt(0);
        }
        __syncthreads();                                 // publish la/lbT[cur]

        Frag a[2];
#pragma unroll
        for (int mi = 0; mi < 2; ++mi) {
            const int row = wave * 32 + mi * 16 + ln;
            a[mi].q[0] = *(const v4u*)&la[cur][row][8 * g];
            a[mi].q[1] = *(const v4u*)&la[cur][row][16 + 8 * g];
        }
#pragma unroll
        for (int j = 0; j < 4; ++j) {
            Frag b;
            b.q[0] = *(const v4u*)&lbT[cur][j * 16 + ln][8 * g];
            b.q[1] = *(const v4u*)&lbT[cur][j * 16 + ln][16 + 8 * g];
#pragma unroll
            for (int mi = 0; mi < 2; ++mi)
                acc[mi][j] = wmma_bf16(a[mi], b, acc[mi][j]);
        }
        __syncthreads();                                 // done with buffer cur
    }

    // Epilogue. C/D layout: lane(g,n) VGPR r -> row r+8g, col n.
#pragma unroll
    for (int mi = 0; mi < 2; ++mi) {
        const int rowBase = tileM + wave * 32 + mi * 16 + 8 * g;
#pragma unroll
        for (int j = 0; j < 4; ++j) {
            if (mode == 0) {
#pragma unroll
                for (int r = 0; r < 8; ++r) {
                    outF[(size_t)(rowBase + r) * N + tileN + j * 16 + ln] =
                        acc[mi][j][r];
                }
            } else {
                const int col   = tileN + j * 16 + ln;
                const int which = col >> 10;          // 0:Q 1:K 2:V
                const int c     = col & 1023;
                const int hh    = c >> 6;
                const int d     = c & 63;
                const int b_    = rowBase >> 11;      // T = 2048
                const int t0    = rowBase & 2047;
                if (which == 2) {
                    // V d-major: 8 consecutive t -> one b128 store
                    Frag pk;
#pragma unroll
                    for (int r = 0; r < 8; ++r) pk.s[r] = f2bf(acc[mi][j][r]);
                    size_t base = (((size_t)b_ * 16 + hh) * 64 + d) * 2048 + t0;
                    *(v4u*)(oVt + base) = pk.q[0];
                } else {
                    u16* o = (which == 0) ? oQ : oK;
#pragma unroll
                    for (int r = 0; r < 8; ++r) {
                        size_t dst =
                            (((size_t)b_ * 16 + hh) * 2048 + t0 + r) * 64 + d;
                        o[dst] = f2bf(acc[mi][j][r]);
                    }
                }
            }
        }
    }
}

// ---------------------------------------------------------------------------
// Flash attention. Q,K: [bh][T][64] bf16; Vt: [bh][64][T] bf16.
// Block = 128 threads = 4 waves; 64 q-rows per block (16 per wave).
// ---------------------------------------------------------------------------
__global__ __launch_bounds__(128)
void flash_attn_kernel(const u16* __restrict__ Q, const u16* __restrict__ Kt,
                       const u16* __restrict__ Vt, u16* __restrict__ Y) {
    const int T = 2048, HD = 64;
    const int bh    = blockIdx.y;
    const int qtile = blockIdx.x;
    const u16* Qp  = Q  + (size_t)bh * T * HD;
    const u16* Kp  = Kt + (size_t)bh * T * HD;
    const u16* Vtp = Vt + (size_t)bh * HD * T;

    const int tid  = threadIdx.x;
    const int wave = tid >> 5;
    const int lane = tid & 31;
    const int g    = lane >> 4;
    const int ln   = lane & 15;
    const int qr   = qtile * 64 + wave * 16;

    __shared__ u16 pT[4][16][32];

    Frag aq0, aq1;
    {
        const u16* qrow = Qp + (size_t)(qr + ln) * HD;
        aq0.q[0] = *(const v4u*)(qrow + 8 * g);
        aq0.q[1] = *(const v4u*)(qrow + 16 + 8 * g);
        aq1.q[0] = *(const v4u*)(qrow + 32 + 8 * g);
        aq1.q[1] = *(const v4u*)(qrow + 48 + 8 * g);
    }

    v8f o[4] = {};
    float mx[8], ls[8];
#pragma unroll
    for (int r = 0; r < 8; ++r) { mx[r] = -1e30f; ls[r] = 0.0f; }

    const float scale = 0.125f;
    const int nkt = (qtile * 64 + 64) >> 5;

    for (int kt = 0; kt < nkt; ++kt) {
        const int kbase = kt * 32;

        v8f s0 = {}, s1 = {};
        {
            const u16* kr0 = Kp + (size_t)(kbase + ln) * HD + 8 * g;
            const u16* kr1 = Kp + (size_t)(kbase + 16 + ln) * HD + 8 * g;
            Frag b00, b01, b10, b11;
            b00.q[0] = *(const v4u*)(kr0);      b00.q[1] = *(const v4u*)(kr0 + 16);
            b01.q[0] = *(const v4u*)(kr0 + 32); b01.q[1] = *(const v4u*)(kr0 + 48);
            b10.q[0] = *(const v4u*)(kr1);      b10.q[1] = *(const v4u*)(kr1 + 16);
            b11.q[0] = *(const v4u*)(kr1 + 32); b11.q[1] = *(const v4u*)(kr1 + 48);
            s0 = wmma_bf16(aq0, b00, s0);
            s0 = wmma_bf16(aq1, b01, s0);
            s1 = wmma_bf16(aq0, b10, s1);
            s1 = wmma_bf16(aq1, b11, s1);
        }

#pragma unroll
        for (int r = 0; r < 8; ++r) {
            int qi = qr + r + 8 * g;
            float v0 = s0[r] * scale;
            float v1 = s1[r] * scale;
            if (kbase + ln > qi)      v0 = -1e30f;
            if (kbase + 16 + ln > qi) v1 = -1e30f;
            s0[r] = v0; s1[r] = v1;
        }

#pragma unroll
        for (int r = 0; r < 8; ++r) {
            float tm = fmaxf(s0[r], s1[r]);
#pragma unroll
            for (int off = 1; off < 16; off <<= 1)
                tm = fmaxf(tm, __shfl_xor(tm, off, 32));
            float mn    = fmaxf(mx[r], tm);
            float alpha = __expf(mx[r] - mn);
            mx[r] = mn;
            float p0 = __expf(s0[r] - mn);
            float p1 = __expf(s1[r] - mn);
            s0[r] = p0; s1[r] = p1;
            float rs = p0 + p1;
#pragma unroll
            for (int off = 1; off < 16; off <<= 1)
                rs += __shfl_xor(rs, off, 32);
            ls[r] = ls[r] * alpha + rs;
            o[0][r] *= alpha; o[1][r] *= alpha;
            o[2][r] *= alpha; o[3][r] *= alpha;
        }

#pragma unroll
        for (int r = 0; r < 8; ++r) {
            pT[wave][r + 8 * g][ln]      = f2bf(s0[r]);
            pT[wave][r + 8 * g][16 + ln] = f2bf(s1[r]);
        }
        Frag ap;
        ap.q[0] = *(const v4u*)&pT[wave][ln][8 * g];
        ap.q[1] = *(const v4u*)&pT[wave][ln][16 + 8 * g];

#pragma unroll
        for (int j = 0; j < 4; ++j) {
            const u16* vr = Vtp + (size_t)(j * 16 + ln) * T + kbase + 8 * g;
            Frag bv;
            bv.q[0] = *(const v4u*)(vr);
            bv.q[1] = *(const v4u*)(vr + 16);
            o[j] = wmma_bf16(ap, bv, o[j]);
        }
    }

    const int b_ = bh >> 4;
    const int hh = bh & 15;
#pragma unroll
    for (int j = 0; j < 4; ++j) {
#pragma unroll
        for (int r = 0; r < 8; ++r) {
            int trow = qr + r + 8 * g;
            float v  = o[j][r] / ls[r];
            size_t idx = ((size_t)b_ * T + trow) * 1024 + hh * 64 + j * 16 + ln;
            Y[idx] = f2bf(v);
        }
    }
}

// ---------------------------------------------------------------------------
// Host-side launcher
// ---------------------------------------------------------------------------
extern "C" void kernel_launch(void* const* d_in, const int* in_sizes, int n_in,
                              void* d_out, int out_size, void* d_ws, size_t ws_size,
                              hipStream_t stream) {
    const int B = 2, T = 2048, C = 1024, H = 16;
    const int M  = B * T;        // 4096
    const int N1 = 3 * C;        // 3072

    const float* X     = (const float*)d_in[0];
    const float* Wqkv  = (const float*)d_in[1];
    const float* Wproj = (const float*)d_in[2];
    float* out = (float*)d_out;

    u16* Xh  = (u16*)d_ws;
    u16* WqT = Xh  + (size_t)M * C;
    u16* WpT = WqT + (size_t)N1 * C;
    u16* Qh  = WpT + (size_t)C * C;
    u16* Kh  = Qh  + (size_t)M * C;
    u16* VtT = Kh  + (size_t)M * C;
    u16* Yh  = VtT + (size_t)M * C;

    {
        int n = M * C;
        f32_to_bf16_kernel<<<(n + 255) / 256, 256, 0, stream>>>(X, Xh, n);
        dim3 blk(32, 8);
        dim3 g1(N1 / 32, C / 32);
        transpose_convert_kernel<<<g1, blk, 0, stream>>>(Wqkv, WqT, C, N1);
        dim3 g2(C / 32, C / 32);
        transpose_convert_kernel<<<g2, blk, 0, stream>>>(Wproj, WpT, C, C);
    }

    {
        dim3 grid(N1 / 64, M / 128);
        gemm_bf16_tn_kernel<<<grid, 128, 0, stream>>>(Xh, WqT, M, N1, C,
                                                      nullptr, Qh, Kh, VtT, 1);
    }

    {
        dim3 grid(T / 64, B * H);
        flash_attn_kernel<<<grid, 128, 0, stream>>>(Qh, Kh, VtT, Yh);
    }

    {
        dim3 grid(C / 64, M / 128);
        gemm_bf16_tn_kernel<<<grid, 128, 0, stream>>>(Yh, WpT, M, C, C,
                                                      out, nullptr, nullptr, nullptr, 0);
    }
}
